// SpatialGRU_8504035246223
// MI455X (gfx1250) — compile-verified, matrix operand order checked by reading, not run on hardware
//
#include <hip/hip_runtime.h>
#include <hip/hip_bf16.h>

// ---------------------------------------------------------------------------
// SpatialGRU on MI455X (gfx1250): anti-diagonal wavefront, per-cell bf16 WMMA
// GEMMs (v_wmma_f32_16x16x32_bf16, f32 accumulate), LDS-staged Q and G,
// column-owned B fragments (no redundant L2 reads), async global->LDS copy
// for the bf16 s-tile (ASYNCcnt path).
// ---------------------------------------------------------------------------

typedef __attribute__((ext_vector_type(16))) __bf16 v16bf;
typedef __attribute__((ext_vector_type(8)))  __bf16 v8bf;
typedef __attribute__((ext_vector_type(8)))  float  v8f;

union AFrag { v16bf v; v8bf h[2]; };

// Problem constants
constexpr int BATCH = 128;     // B
constexpr int UH    = 64;      // hidden
constexpr int CCH   = 64;      // channels
constexpr int L     = 64;      // L1 == L2
constexpr int KQ    = 3 * UH + CCH;   // 256 (K of both GEMMs)
constexpr int N1    = 7 * UH;         // 448
constexpr int NT1   = N1 / 16;        // 28
constexpr int KT    = KQ / 32;        // 8
constexpr int NT2   = UH / 16;        // 4

// Workspace layout (bytes)
constexpr size_t H_ELEMS  = (size_t)(L + 1) * (L + 1) * BATCH * UH;    // ghost grid
constexpr size_t XT_ELEMS = (size_t)L * L * BATCH * CCH;
constexpr size_t WP_ELEMS = (size_t)KT * NT1 * 32 * 16;                // 114688
constexpr size_t U2_ELEMS = (size_t)KT * NT2 * 32 * 16;                // 16384
constexpr size_t H_BYTES  = H_ELEMS * sizeof(float);
constexpr size_t XT_BYTES = XT_ELEMS * sizeof(unsigned short);
constexpr size_t WP_BYTES = WP_ELEMS * sizeof(unsigned short);

constexpr int CELL_STRIDE = BATCH * UH;      // 8192 floats per grid cell
constexpr unsigned SMEM_BYTES = BATCH * KQ * 2 + BATCH * N1 * 4;   // 294912

__device__ __forceinline__ unsigned short f2bf(float f) {
    union { float f; unsigned u; } v; v.f = f;
    unsigned r = v.u + 0x7FFFu + ((v.u >> 16) & 1u);
    return (unsigned short)(r >> 16);
}

__device__ __forceinline__ void store_bf4(unsigned short* p, float4 f) {
    union { unsigned short s[4]; uint2 u; } r;
    r.s[0] = f2bf(f.x); r.s[1] = f2bf(f.y); r.s[2] = f2bf(f.z); r.s[3] = f2bf(f.w);
    *(uint2*)p = r.u;
}

// 16B async DMA: global -> LDS, tracked by ASYNCcnt (gfx1250)
__device__ __forceinline__ void async_copy16(unsigned lds_off, unsigned long long gaddr) {
    asm volatile("global_load_async_to_lds_b128 %0, %1, off"
                 :: "v"(lds_off), "v"(gaddr) : "memory");
}
__device__ __forceinline__ void wait_async0() {
    asm volatile("s_wait_asynccnt 0x0" ::: "memory");
}

// --------------------------- prep kernels ----------------------------------

// Pack W (256x448 f32 row-major) into WMMA B-fragment blocked bf16:
// Wp[((kt*NT1+nt)*32 + lane)*16 + t]; lane&15 -> N, lane>>4 selects K halves.
__global__ void pack_W_kernel(const float* __restrict__ W, unsigned short* __restrict__ Wp) {
    int idx = blockIdx.x * 256 + threadIdx.x;
    if (idx >= (int)WP_ELEMS) return;
    int t = idx & 15, lane = (idx >> 4) & 31, rem = idx >> 9;
    int nt = rem % NT1, kt = rem / NT1;
    int n = lane & 15, hi = lane >> 4;
    int koff = (t < 8) ? (hi * 8 + t) : (16 + hi * 8 + (t - 8));
    int K = kt * 32 + koff, N = nt * 16 + n;
    Wp[idx] = f2bf(W[K * N1 + N]);
}

// Pack [Umat (192x64); Wij (64x64)] the same way.
__global__ void pack_U2_kernel(const float* __restrict__ Umat, const float* __restrict__ Wij,
                               unsigned short* __restrict__ U2p) {
    int idx = blockIdx.x * 256 + threadIdx.x;
    if (idx >= (int)U2_ELEMS) return;
    int t = idx & 15, lane = (idx >> 4) & 31, rem = idx >> 9;
    int nt = rem & 3, kt = rem >> 2;
    int n = lane & 15, hi = lane >> 4;
    int koff = (t < 8) ? (hi * 8 + t) : (16 + hi * 8 + (t - 8));
    int K = kt * 32 + koff, N = nt * 16 + n;
    float v = (K < 3 * UH) ? Umat[K * UH + N] : Wij[(K - 3 * UH) * UH + N];
    U2p[idx] = f2bf(v);
}

// x (B,C,L,L) f32 -> xT (i,j,b,c) bf16
__global__ void transpose_x_kernel(const float* __restrict__ x, unsigned short* __restrict__ xT) {
    size_t o = (size_t)blockIdx.x * 256 + threadIdx.x;
    if (o >= XT_ELEMS) return;
    int c = (int)(o & 63);
    int b = (int)((o >> 6) & 127);
    int j = (int)((o >> 13) & 63);
    int i = (int)(o >> 19);
    xT[o] = f2bf(x[((size_t)(b * CCH + c) << 12) + (i << 6) + j]);
}

// zero ghost row 0 and ghost column 0 of the (65x65) h grid
__global__ void zero_borders_kernel(float* __restrict__ Hg) {
    int t = blockIdx.x * 256 + threadIdx.x;
    const int RB = (L + 1) * CELL_STRIDE;            // ghost row 0: cells 0..64 contiguous
    if (t < RB) {
        Hg[t] = 0.0f;
    } else if (t < RB + L * CELL_STRIDE) {
        int k = t - RB;
        int i = (k >> 13) + 1;                       // ghost rows 1..64, col 0
        int off = k & (CELL_STRIDE - 1);
        Hg[(size_t)i * (L + 1) * CELL_STRIDE + off] = 0.0f;
    }
}

__global__ void copy_out_kernel(const float* __restrict__ Hg, float* __restrict__ out) {
    int t = blockIdx.x * 256 + threadIdx.x;
    if (t < CELL_STRIDE)
        out[t] = Hg[((size_t)L * (L + 1) + L) * CELL_STRIDE + t];
}

// --------------------------- per-cell kernel -------------------------------

__global__ __launch_bounds__(256, 1)
void cell_kernel(float* __restrict__ Hg, const unsigned short* __restrict__ xT,
                 const unsigned short* __restrict__ Wp, const unsigned short* __restrict__ U2p,
                 const float* __restrict__ bias, int dgn, int ilo) {
    extern __shared__ char smem_raw[];
    unsigned short* Qs = (unsigned short*)smem_raw;              // 128x256 bf16 (offset 0)
    float* Gs = (float*)(smem_raw + BATCH * KQ * 2);             // 128x448 f32

    const int tid = threadIdx.x;
    const int i = ilo + blockIdx.x;
    const int j = dgn - i;

    const float* h_top_p  = Hg + ((size_t)i       * (L + 1) + (j + 1)) * CELL_STRIDE; // h(i-1,j)
    const float* h_left_p = Hg + ((size_t)(i + 1) * (L + 1) + j)       * CELL_STRIDE; // h(i,j-1)
    const float* h_diag_p = Hg + ((size_t)i       * (L + 1) + j)       * CELL_STRIDE; // h(i-1,j-1)
    float*       h_out_p  = Hg + ((size_t)(i + 1) * (L + 1) + (j + 1)) * CELL_STRIDE;
    const unsigned short* s_p = xT + ((size_t)(i * L + j)) * BATCH * CCH;

    const int wave = tid >> 5, lane = tid & 31;
    const int ln = lane & 15, hi = lane >> 4;

    // ---- Phase 1a: async DMA the bf16 s-tile (16KB) into Q cols 192..255 ----
    // Q row b occupies 512B; s bytes for row b land at b*512 + 384 .. +511.
    {
        unsigned long long sbase = (unsigned long long)(uintptr_t)s_p;
        #pragma unroll
        for (int it = 0; it < 4; ++it) {
            unsigned t = (unsigned)((wave * 4 + it) * 512 + lane * 16);   // linear byte off in s
            unsigned lds_off = ((t >> 7) * 512) + 384 + (t & 127);        // Q region starts at 0
            async_copy16(lds_off, sbase + t);
        }
    }

    // ---- Phase 1b: h sections f32 -> bf16 into Q cols 0..191 (overlaps DMA) ----
    for (int e = tid; e < BATCH * UH / 4; e += 256) {            // 2048 float4 per section
        int b  = e >> 4;
        int u0 = (e & 15) * 4;
        float4 ft = ((const float4*)h_top_p)[e];
        float4 fl = ((const float4*)h_left_p)[e];
        float4 fd = ((const float4*)h_diag_p)[e];
        unsigned short* q = Qs + b * KQ;
        store_bf4(q + 0   + u0, ft);
        store_bf4(q + 64  + u0, fl);
        store_bf4(q + 128 + u0, fd);
    }
    wait_async0();
    __syncthreads();

    // ---- Phase 2: G = Q @ W + bias  (M=128, N=448, K=256) ----
    // Each wave OWNS a set of N-columns: B fragments read exactly once per WG.
    {
        int cnt1   = (wave < 4) ? 4 : 3;                      // 4,4,4,4,3,3,3,3 columns
        int start1 = (wave < 4) ? wave * 4 : 16 + (wave - 4) * 3;
        for (int qc = 0; qc < cnt1; ++qc) {
            int nt = start1 + qc;
            AFrag bfrg[KT];
            #pragma unroll
            for (int kt = 0; kt < KT; ++kt)
                bfrg[kt].v = *(const v16bf*)(Wp + ((size_t)(kt * NT1 + nt) * 32 + lane) * 16);
            int col = nt * 16 + ln;
            float bv = bias[col];
            for (int mt = 0; mt < 8; ++mt) {
                const unsigned short* qb = Qs + (mt * 16 + ln) * KQ + hi * 8;
                v8f c = {0.f, 0.f, 0.f, 0.f, 0.f, 0.f, 0.f, 0.f};
                #pragma unroll
                for (int kt = 0; kt < KT; ++kt) {
                    AFrag a;
                    a.h[0] = *(const v8bf*)(qb + kt * 32);
                    a.h[1] = *(const v8bf*)(qb + kt * 32 + 16);
                    c = __builtin_amdgcn_wmma_f32_16x16x32_bf16(false, a.v, false, bfrg[kt].v,
                                                                (short)0, c, false, false);
                }
                float* g = Gs + (mt * 16 + hi * 8) * N1 + col;
                #pragma unroll
                for (int v = 0; v < 8; ++v) g[v * N1] = c[v] + bv;
            }
        }
    }
    __syncthreads();

    // ---- Phase 3: sigmoid/softmax; build Q2 = [r*hl | r*ht | r*hd | s] ----
    for (int it = 0; it < 32; ++it) {
        int idx = it * 256 + tid;          // 8192 = 128 x 64
        int b = idx >> 6, u = idx & 63;
        float* grow = Gs + b * N1;
        float z0 = grow[192 + u], z1 = grow[256 + u], z2 = grow[320 + u], z3 = grow[384 + u];
        float m = fmaxf(fmaxf(z0, z1), fmaxf(z2, z3));
        float e0 = __expf(z0 - m), e1 = __expf(z1 - m), e2 = __expf(z2 - m), e3 = __expf(z3 - m);
        float inv = 1.0f / (e0 + e1 + e2 + e3);
        float zi = e0 * inv, zl = e1 * inv, zt = e2 * inv, zd = e3 * inv;
        float r0 = 1.0f / (1.0f + __expf(-grow[u]));
        float r1 = 1.0f / (1.0f + __expf(-grow[64 + u]));
        float r2 = 1.0f / (1.0f + __expf(-grow[128 + u]));
        float hl = h_left_p[b * UH + u];
        float ht = h_top_p[b * UH + u];
        float hd = h_diag_p[b * UH + u];
        unsigned short* q = Qs + b * KQ;   // overwrite reset-gated inputs (order: left, top, diag)
        q[u]        = f2bf(r0 * hl);
        q[64 + u]   = f2bf(r1 * ht);
        q[128 + u]  = f2bf(r2 * hd);
        grow[u]      = zl * hl + zt * ht + zd * hd;   // stash convex partial
        grow[64 + u] = zi;                            // stash z_i
    }
    __syncthreads();

    // ---- Phase 4: hij_ = tanh(Q2 @ [U;Wij] + bij); h = part + zi * hij_ ----
    // 8 waves over 4 columns: 2 waves per column, 4 M-tiles each (32 wmma/wave).
    {
        int nt  = wave >> 1;
        int mlo = (wave & 1) * 4;
        AFrag bfrg[KT];
        #pragma unroll
        for (int kt = 0; kt < KT; ++kt)
            bfrg[kt].v = *(const v16bf*)(U2p + ((size_t)(kt * NT2 + nt) * 32 + lane) * 16);
        int col = nt * 16 + ln;            // hidden index u'
        float bv = bias[N1 + col];
        for (int mt = mlo; mt < mlo + 4; ++mt) {
            const unsigned short* qb = Qs + (mt * 16 + ln) * KQ + hi * 8;
            v8f c = {0.f, 0.f, 0.f, 0.f, 0.f, 0.f, 0.f, 0.f};
            #pragma unroll
            for (int kt = 0; kt < KT; ++kt) {
                AFrag a;
                a.h[0] = *(const v8bf*)(qb + kt * 32);
                a.h[1] = *(const v8bf*)(qb + kt * 32 + 16);
                c = __builtin_amdgcn_wmma_f32_16x16x32_bf16(false, a.v, false, bfrg[kt].v,
                                                            (short)0, c, false, false);
            }
            #pragma unroll
            for (int v = 0; v < 8; ++v) {
                int m = mt * 16 + hi * 8 + v;
                float hij_ = tanhf(c[v] + bv);
                float part = Gs[m * N1 + col];
                float ziv  = Gs[m * N1 + 64 + col];
                h_out_p[m * UH + col] = part + ziv * hij_;
            }
        }
    }
}

// ------------------------------- host --------------------------------------

extern "C" void kernel_launch(void* const* d_in, const int* in_sizes, int n_in,
                              void* d_out, int out_size, void* d_ws, size_t ws_size,
                              hipStream_t stream) {
    const float* x    = (const float*)d_in[0];
    const float* W    = (const float*)d_in[1];
    const float* Umat = (const float*)d_in[2];
    const float* Wij  = (const float*)d_in[3];
    const float* bias = (const float*)d_in[4];
    float* out = (float*)d_out;

    char* ws = (char*)d_ws;
    float*          Hg  = (float*)ws;
    unsigned short* xT  = (unsigned short*)(ws + H_BYTES);
    unsigned short* Wp  = (unsigned short*)(ws + H_BYTES + XT_BYTES);
    unsigned short* U2p = (unsigned short*)(ws + H_BYTES + XT_BYTES + WP_BYTES);

    (void)hipFuncSetAttribute((const void*)cell_kernel,
                              hipFuncAttributeMaxDynamicSharedMemorySize, (int)SMEM_BYTES);

    pack_W_kernel<<<((int)WP_ELEMS + 255) / 256, 256, 0, stream>>>(W, Wp);
    pack_U2_kernel<<<((int)U2_ELEMS + 255) / 256, 256, 0, stream>>>(Umat, Wij, U2p);
    transpose_x_kernel<<<(int)((XT_ELEMS + 255) / 256), 256, 0, stream>>>(x, xT);
    {
        int n = (2 * L + 1) * CELL_STRIDE;  // 129 border cells
        zero_borders_kernel<<<(n + 255) / 256, 256, 0, stream>>>(Hg);
    }

    for (int d = 0; d < 2 * L - 1; ++d) {
        int ilo = d > (L - 1) ? d - (L - 1) : 0;
        int ihi = d < (L - 1) ? d : (L - 1);
        int ncells = ihi - ilo + 1;
        cell_kernel<<<ncells, 256, SMEM_BYTES, stream>>>(Hg, xT, Wp, U2p, bias, d, ilo);
    }

    copy_out_kernel<<<(CELL_STRIDE + 255) / 256, 256, 0, stream>>>(Hg, out);
}
